// GCN_44676249813667
// MI455X (gfx1250) — matrix operand order repaired and checked by reading, not compiled, not run on hardware
//
#include <hip/hip_runtime.h>
#include <cstdint>

typedef __bf16 bf16;
typedef __attribute__((ext_vector_type(16))) __bf16 v16bf;
typedef __attribute__((ext_vector_type(8)))  __bf16 v8bf;
typedef __attribute__((ext_vector_type(8)))  float  v8f;

#define NN 50000
#define KK 8
#define PP 512
#define CC 64
#define HH 128
#define EE 1600000L
#define QQ 500000L
#define GG 64

// ---------------------------------------------------------------- helpers ---
__device__ __forceinline__ float gelu_f(float x) {
    // exact erf GELU (matches torch F.gelu default)
    return 0.5f * x * (1.0f + erff(x * 0.7071067811865476f));
}

// A-fragment (16x32 bf16, MxK): lane L holds row L&15; K-chunks per ISA layout.
__device__ __forceinline__ v16bf load_afrag(const bf16* A, int lda, int kb, int lane) {
    const bf16* p = A + (size_t)(lane & 15) * lda + kb + ((lane >> 4) << 3);
    v8bf lo = *(const v8bf*)p;
    v8bf hi = *(const v8bf*)(p + 16);
    return __builtin_shufflevector(lo, hi, 0,1,2,3,4,5,6,7,8,9,10,11,12,13,14,15);
}

// B-fragment (32x16 bf16, KxN) from row-major W (OxK): lane L holds col n=L&15,
// K-chunks per ISA layout (lanes 0-15: K 0..7,16..23; lanes 16-31: K 8..15,24..31).
__device__ __forceinline__ v16bf load_bfrag(const bf16* W, int ldw, int n0, int kb, int lane) {
    const bf16* p = W + (size_t)(n0 + (lane & 15)) * ldw + kb + ((lane >> 4) << 3);
    v8bf lo = *(const v8bf*)p;
    v8bf hi = *(const v8bf*)(p + 16);
    return __builtin_shufflevector(lo, hi, 0,1,2,3,4,5,6,7,8,9,10,11,12,13,14,15);
}

// acc += A(16xK) * W^T tile (n0..n0+15) over full K, bf16 WMMA, fp32 accum.
__device__ __forceinline__ v8f gemm_acc(const bf16* A, int lda, const bf16* W, int ldw,
                                        int n0, int Kdim, v8f acc, int lane) {
    for (int kb = 0; kb < Kdim; kb += 32) {
        v16bf a = load_afrag(A, lda, kb, lane);
        v16bf b = load_bfrag(W, ldw, n0, kb, lane);
        acc = __builtin_amdgcn_wmma_f32_16x16x32_bf16(false, a, false, b,
                                                      (short)0, acc, false, false);
    }
    return acc;
}

__device__ __forceinline__ v8f bias_bcast(const float* b, int n0, int lane) {
    float v = b[n0 + (lane & 15)];
    v8f c;
#pragma unroll
    for (int i = 0; i < 8; ++i) c[i] = v;
    return c;
}

// --------------------------------------------------------------- utility ----
__global__ void k_cvt_bf16(const float* __restrict__ s, bf16* __restrict__ d, int n) {
    int i = blockIdx.x * blockDim.x + threadIdx.x;
    if (i < n) d[i] = (bf16)s[i];
}

__global__ void k_fill(float* __restrict__ p, long n, float v) {
    long i = (long)blockIdx.x * blockDim.x + threadIdx.x;
    if (i < n) p[i] = v;
}

__global__ void k_copy(const float* __restrict__ s, float* __restrict__ d, long n) {
    long i = (long)blockIdx.x * blockDim.x + threadIdx.x;
    if (i < n) d[i] = s[i];
}

// ------------------------------------------------- 1) embedder per-point ----
// rows = N*K (node-major, 8 points contiguous). Wave tile = 16 rows = 2 nodes.
// y = gelu(mlp(x)); hsum[node] = sum over the 8 point rows (falls out of C layout).
#define EMB_WAVES 2
__global__ __launch_bounds__(EMB_WAVES * 32)
void k_embed(const float* __restrict__ x,
             const bf16* __restrict__ Wi, const bf16* __restrict__ Wh,
             const bf16* __restrict__ Wo, const bf16* __restrict__ Wp,
             const float* __restrict__ bi, const float* __restrict__ bh,
             const float* __restrict__ bo, const float* __restrict__ bp,
             bf16* __restrict__ hsum) {
    __shared__ __align__(16) bf16 lds[EMB_WAVES * (16 * PP + 2 * 16 * HH)];
    const int wave = threadIdx.x >> 5, lane = threadIdx.x & 31;
    bf16* A  = lds + wave * (16 * PP + 2 * 16 * HH);
    bf16* S1 = A + 16 * PP;
    bf16* S2 = S1 + 16 * HH;
    const int tile = blockIdx.x * EMB_WAVES + wave;
    if (tile >= (NN * KK) / 16) return;
    const size_t row0 = (size_t)tile * 16;

    // stage x tile (contiguous 16x512 fp32) -> bf16 LDS
    const float* xp = x + row0 * PP;
    for (int i = lane * 4; i < 16 * PP; i += 32 * 4) {
        float4 v = *(const float4*)(xp + i);
        A[i + 0] = (bf16)v.x; A[i + 1] = (bf16)v.y;
        A[i + 2] = (bf16)v.z; A[i + 3] = (bf16)v.w;
    }
    const int n = lane & 15, mo = (lane >> 4) * 8;
    v8f skip[8];
#pragma unroll
    for (int nt = 0; nt < 8; ++nt) {                    // t1 = gelu(x@Wi^T+bi); skip = x@Wp^T+bp
        v8f acc = bias_bcast(bi, nt * 16, lane);
        acc = gemm_acc(A, PP, Wi, PP, nt * 16, PP, acc, lane);
#pragma unroll
        for (int r = 0; r < 8; ++r) S1[(mo + r) * HH + nt * 16 + n] = (bf16)gelu_f(acc[r]);
        skip[nt] = gemm_acc(A, PP, Wp, PP, nt * 16, PP, bias_bcast(bp, nt * 16, lane), lane);
    }
#pragma unroll
    for (int nt = 0; nt < 8; ++nt) {                    // u = skip + gelu(t1@Wh^T+bh)
        v8f acc = bias_bcast(bh, nt * 16, lane);
        acc = gemm_acc(S1, HH, Wh, HH, nt * 16, HH, acc, lane);
#pragma unroll
        for (int r = 0; r < 8; ++r)
            S2[(mo + r) * HH + nt * 16 + n] = (bf16)(skip[nt][r] + gelu_f(acc[r]));
    }
    const size_t node = row0 / KK + (lane >> 4);        // lower half-wave: node0, upper: node1
    for (int nt = 0; nt < 32; ++nt) {                   // y = gelu(u@Wo^T+bo); reduce 8 rows
        v8f acc = bias_bcast(bo, nt * 16, lane);
        acc = gemm_acc(S2, HH, Wo, HH, nt * 16, HH, acc, lane);
        float s = 0.f;
#pragma unroll
        for (int r = 0; r < 8; ++r) s += gelu_f(acc[r]);
        hsum[node * PP + nt * 16 + n] = (bf16)s;
    }
}

// ---------------------------------------------------------- 2) rho MLP ------
#define RHO_WAVES 4
__global__ __launch_bounds__(RHO_WAVES * 32)
void k_rho(const bf16* __restrict__ hsum,
           const bf16* __restrict__ Wi, const bf16* __restrict__ Wh,
           const bf16* __restrict__ Wo, const bf16* __restrict__ Wp,
           const float* __restrict__ bi, const float* __restrict__ bh,
           const float* __restrict__ bo, const float* __restrict__ bp,
           float* __restrict__ z) {
    __shared__ __align__(16) bf16 lds[RHO_WAVES * 2 * 16 * HH];
    const int wave = threadIdx.x >> 5, lane = threadIdx.x & 31;
    const int tile = blockIdx.x * RHO_WAVES + wave;
    if (tile >= NN / 16) return;
    bf16* S1 = lds + wave * 2 * 16 * HH;
    bf16* S2 = S1 + 16 * HH;
    const size_t row0 = (size_t)tile * 16;
    const bf16* A = hsum + row0 * PP;                   // A frags straight from global bf16
    const int n = lane & 15, mo = (lane >> 4) * 8;
    v8f skip[8];
#pragma unroll
    for (int nt = 0; nt < 8; ++nt) {
        v8f acc = bias_bcast(bi, nt * 16, lane);
        acc = gemm_acc(A, PP, Wi, PP, nt * 16, PP, acc, lane);
#pragma unroll
        for (int r = 0; r < 8; ++r) S1[(mo + r) * HH + nt * 16 + n] = (bf16)gelu_f(acc[r]);
        skip[nt] = gemm_acc(A, PP, Wp, PP, nt * 16, PP, bias_bcast(bp, nt * 16, lane), lane);
    }
#pragma unroll
    for (int nt = 0; nt < 8; ++nt) {
        v8f acc = bias_bcast(bh, nt * 16, lane);
        acc = gemm_acc(S1, HH, Wh, HH, nt * 16, HH, acc, lane);
#pragma unroll
        for (int r = 0; r < 8; ++r)
            S2[(mo + r) * HH + nt * 16 + n] = (bf16)(skip[nt][r] + gelu_f(acc[r]));
    }
#pragma unroll
    for (int nt = 0; nt < 4; ++nt) {                    // out width C=64
        v8f acc = bias_bcast(bo, nt * 16, lane);
        acc = gemm_acc(S2, HH, Wo, HH, nt * 16, HH, acc, lane);
#pragma unroll
        for (int r = 0; r < 8; ++r)
            z[(row0 + mo + r) * CC + nt * 16 + n] = acc[r];
    }
}

// ------------------------------------------------ 3) SAGE conv (scatter) ----
__global__ void k_edges(const int* __restrict__ ei, const float* __restrict__ z,
                        float* __restrict__ agg, float* __restrict__ cnt) {
    long gid = (long)blockIdx.x * blockDim.x + threadIdx.x;
    if (gid >= EE * CC) return;
    long e = gid >> 6;
    int f = (int)(gid & 63);
    int s = ei[e], d = ei[EE + e];
    atomicAdd(&agg[(size_t)d * CC + f], z[(size_t)s * CC + f]);
    if (f == 0) atomicAdd(&cnt[d], 1.0f);
}

#define SG_WAVES 8
__global__ __launch_bounds__(SG_WAVES * 32)
void k_sage(const float* __restrict__ zin, const float* __restrict__ agg,
            const float* __restrict__ cnt, const bf16* __restrict__ Wl,
            const bf16* __restrict__ Wr, const float* __restrict__ bl,
            float* __restrict__ zout) {
    __shared__ __align__(16) bf16 lds[SG_WAVES * 2 * 16 * CC];
    const int wave = threadIdx.x >> 5, lane = threadIdx.x & 31;
    const int tile = blockIdx.x * SG_WAVES + wave;
    if (tile >= NN / 16) return;
    bf16* SA = lds + wave * 2 * 16 * CC;
    bf16* SZ = SA + 16 * CC;
    const size_t row0 = (size_t)tile * 16;
    for (int i = lane; i < 16 * CC; i += 32) {
        int m = i >> 6, f = i & 63;
        size_t r = row0 + m;
        float c = cnt[r]; c = c < 1.f ? 1.f : c;
        SA[i] = (bf16)(agg[r * CC + f] / c);
        SZ[i] = (bf16)zin[r * CC + f];
    }
    const int n = lane & 15, mo = (lane >> 4) * 8;
#pragma unroll
    for (int nt = 0; nt < 4; ++nt) {
        v8f acc = bias_bcast(bl, nt * 16, lane);
        acc = gemm_acc(SA, CC, Wl, CC, nt * 16, CC, acc, lane);   // agg@Wl^T + bl
        acc = gemm_acc(SZ, CC, Wr, CC, nt * 16, CC, acc, lane);   // + z@Wr^T
#pragma unroll
        for (int r = 0; r < 8; ++r) {
            size_t m = row0 + mo + r;
            zout[m * CC + nt * 16 + n] = zin[m * CC + nt * 16 + n] + gelu_f(acc[r]);
        }
    }
}

// ---------------------------------------------------- 4) graph max-pool -----
__global__ void k_gmax(const float* __restrict__ z, const int* __restrict__ batch,
                       float* __restrict__ ge) {
    long gid = (long)blockIdx.x * blockDim.x + threadIdx.x;
    if (gid >= (long)NN * CC) return;
    long node = gid >> 6;
    int f = (int)(gid & 63);
    int g = batch[node];
    int iv = __float_as_int(z[gid]);
    int* addr = (int*)&ge[(size_t)g * CC + f];
    if (iv >= 0) atomicMax(addr, iv);                       // float-max via int ordering
    else         atomicMin((unsigned int*)addr, (unsigned int)iv);
}

// ------------------------------------------------- 5) xy output gather ------
__global__ void k_xy(const float* __restrict__ z, const float* __restrict__ ge,
                     const int* __restrict__ batch, const int* __restrict__ srci,
                     const int* __restrict__ dsti, float* __restrict__ out) {
    long gid = (long)blockIdx.x * blockDim.x + threadIdx.x;
    if (gid >= QQ * 192) return;
    long q = gid / 192;
    int j = (int)(gid - q * 192);
    float v;
    if (j < 64)        v = z[(size_t)srci[q] * CC + j];
    else if (j < 128)  v = z[(size_t)dsti[q] * CC + (j - 64)];
    else               v = ge[(size_t)batch[srci[q]] * CC + (j - 128)];
    out[gid] = v;
}

// -------------------------------------------------------- 6) head MLPs ------
#define HD_WAVES 2
__global__ __launch_bounds__(HD_WAVES * 32)
void k_head(const float* __restrict__ z, const float* __restrict__ ge,
            const int* __restrict__ batch, const int* __restrict__ srci,
            const int* __restrict__ dsti, int swap,
            const bf16* __restrict__ W1i, const bf16* __restrict__ W1h,
            const bf16* __restrict__ W1o, const bf16* __restrict__ W1p,
            const bf16* __restrict__ W2i, const bf16* __restrict__ W2h,
            const float* __restrict__ b1i, const float* __restrict__ b1h,
            const float* __restrict__ b1o, const float* __restrict__ b1p,
            const float* __restrict__ b2i, const float* __restrict__ b2h,
            const float* __restrict__ w2o, const float* __restrict__ b2o,
            float* __restrict__ out) {
    __shared__ __align__(16) bf16 lds[HD_WAVES * (16 * 192 + 3 * 16 * HH)];
    __shared__ int idl[HD_WAVES * 48];
    const int wave = threadIdx.x >> 5, lane = threadIdx.x & 31;
    const int tile = blockIdx.x * HD_WAVES + wave;
    if (tile >= QQ / 16) return;
    bf16* SQ = lds + wave * (16 * 192 + 3 * 16 * HH);
    bf16* S1 = SQ + 16 * 192;
    bf16* S2 = S1 + 16 * HH;
    bf16* S3 = S2 + 16 * HH;
    int* id = idl + wave * 48;
    const size_t q0 = (size_t)tile * 16;
    if (lane < 16) {
        int s = srci[q0 + lane], d = dsti[q0 + lane];
        id[lane]      = swap ? d : s;
        id[16 + lane] = swap ? s : d;
        id[32 + lane] = batch[s];                       // ge keyed by batch[src] in both heads
    }
    for (int i = lane; i < 16 * 192; i += 32) {         // gather [za | zb | ge] -> bf16 LDS
        int m = i / 192, j = i - m * 192;
        float v;
        if (j < 64)        v = z[(size_t)id[m] * CC + j];
        else if (j < 128)  v = z[(size_t)id[16 + m] * CC + (j - 64)];
        else               v = ge[(size_t)id[32 + m] * CC + (j - 128)];
        SQ[i] = (bf16)v;
    }
    const int n = lane & 15, mo = (lane >> 4) * 8;
    v8f skip[8];
#pragma unroll
    for (int nt = 0; nt < 8; ++nt) {                    // h1 in + proj (K=192)
        v8f acc = bias_bcast(b1i, nt * 16, lane);
        acc = gemm_acc(SQ, 192, W1i, 192, nt * 16, 192, acc, lane);
#pragma unroll
        for (int r = 0; r < 8; ++r) S1[(mo + r) * HH + nt * 16 + n] = (bf16)gelu_f(acc[r]);
        skip[nt] = gemm_acc(SQ, 192, W1p, 192, nt * 16, 192, bias_bcast(b1p, nt * 16, lane), lane);
    }
#pragma unroll
    for (int nt = 0; nt < 8; ++nt) {                    // h1 hid + skip
        v8f acc = bias_bcast(b1h, nt * 16, lane);
        acc = gemm_acc(S1, HH, W1h, HH, nt * 16, HH, acc, lane);
#pragma unroll
        for (int r = 0; r < 8; ++r)
            S2[(mo + r) * HH + nt * 16 + n] = (bf16)(skip[nt][r] + gelu_f(acc[r]));
    }
#pragma unroll
    for (int nt = 0; nt < 8; ++nt) {                    // h1 out -> relu -> S3
        v8f acc = bias_bcast(b1o, nt * 16, lane);
        acc = gemm_acc(S2, HH, W1o, HH, nt * 16, HH, acc, lane);
#pragma unroll
        for (int r = 0; r < 8; ++r) {
            float t = acc[r]; t = t > 0.f ? t : 0.f;
            S3[(mo + r) * HH + nt * 16 + n] = (bf16)t;
        }
    }
#pragma unroll
    for (int nt = 0; nt < 8; ++nt) {                    // h2 in (reuse S1)
        v8f acc = bias_bcast(b2i, nt * 16, lane);
        acc = gemm_acc(S3, HH, W2i, HH, nt * 16, HH, acc, lane);
#pragma unroll
        for (int r = 0; r < 8; ++r) S1[(mo + r) * HH + nt * 16 + n] = (bf16)gelu_f(acc[r]);
    }
#pragma unroll
    for (int nt = 0; nt < 8; ++nt) {                    // h2 hid + identity skip (reuse S2)
        v8f acc = bias_bcast(b2h, nt * 16, lane);
        acc = gemm_acc(S1, HH, W2h, HH, nt * 16, HH, acc, lane);
#pragma unroll
        for (int r = 0; r < 8; ++r) {
            int off = (mo + r) * HH + nt * 16 + n;
            S2[off] = (bf16)((float)S3[off] + gelu_f(acc[r]));
        }
    }
    if (lane < 16) {                                    // final 128 -> 1 dot
        float s = b2o[0];
        for (int j = 0; j < HH; ++j) s += (float)S2[lane * HH + j] * w2o[j];
        out[q0 + lane] = s;
    }
}

// ----------------------------------------------------------------- host -----
extern "C" void kernel_launch(void* const* d_in, const int* in_sizes, int n_in,
                              void* d_out, int out_size, void* d_ws, size_t ws_size,
                              hipStream_t stream) {
    const float* x       = (const float*)d_in[0];
    const int*   batch   = (const int*)d_in[1];
    const int*   ei      = (const int*)d_in[2];
    const int*   srci    = (const int*)d_in[3];
    const int*   dsti    = (const int*)d_in[4];
    const float* em_in_W = (const float*)d_in[5];  const float* em_in_b  = (const float*)d_in[6];
    const float* em_hid_W= (const float*)d_in[7];  const float* em_hid_b = (const float*)d_in[8];
    const float* em_out_W= (const float*)d_in[9];  const float* em_out_b = (const float*)d_in[10];
    const float* em_pj_W = (const float*)d_in[11]; const float* em_pj_b  = (const float*)d_in[12];
    const float* rh_in_W = (const float*)d_in[13]; const float* rh_in_b  = (const float*)d_in[14];
    const float* rh_hid_W= (const float*)d_in[15]; const float* rh_hid_b = (const float*)d_in[16];
    const float* rh_out_W= (const float*)d_in[17]; const float* rh_out_b = (const float*)d_in[18];
    const float* rh_pj_W = (const float*)d_in[19]; const float* rh_pj_b  = (const float*)d_in[20];
    const float* cllW[3] = {(const float*)d_in[21], (const float*)d_in[24], (const float*)d_in[27]};
    const float* cllB[3] = {(const float*)d_in[22], (const float*)d_in[25], (const float*)d_in[28]};
    const float* clrW[3] = {(const float*)d_in[23], (const float*)d_in[26], (const float*)d_in[29]};
    const float* h1_in_W = (const float*)d_in[30]; const float* h1_in_b  = (const float*)d_in[31];
    const float* h1_hid_W= (const float*)d_in[32]; const float* h1_hid_b = (const float*)d_in[33];
    const float* h1_out_W= (const float*)d_in[34]; const float* h1_out_b = (const float*)d_in[35];
    const float* h1_pj_W = (const float*)d_in[36]; const float* h1_pj_b  = (const float*)d_in[37];
    const float* h2_in_W = (const float*)d_in[38]; const float* h2_in_b  = (const float*)d_in[39];
    const float* h2_hid_W= (const float*)d_in[40]; const float* h2_hid_b = (const float*)d_in[41];
    const float* h2_out_W= (const float*)d_in[42]; const float* h2_out_b = (const float*)d_in[43];

    // workspace partition
    uintptr_t wp = (uintptr_t)d_ws;
    auto take = [&](size_t bytes) -> void* {
        void* r = (void*)wp; wp += (bytes + 255) & ~(size_t)255; return r;
    };
    bf16*  arena = (bf16*)take(2u << 20);                  // bf16 weight arena (~1 MB used)
    bf16*  hsumb = (bf16*)take((size_t)NN * PP * 2);
    float* z0    = (float*)take((size_t)NN * CC * 4);
    float* z1    = (float*)take((size_t)NN * CC * 4);
    float* agg   = (float*)take((size_t)NN * CC * 4);
    float* cnt   = (float*)take((size_t)NN * 4);
    float* ge    = (float*)take((size_t)GG * CC * 4);

    // convert weights fp32 -> bf16 arena
    bf16* ap = arena;
    auto cvt = [&](const float* s, int nel) -> bf16* {
        bf16* d = ap; ap += (nel + 127) & ~127;
        k_cvt_bf16<<<(nel + 255) / 256, 256, 0, stream>>>(s, d, nel);
        return d;
    };
    bf16 *Wem_i = cvt(em_in_W, HH * PP), *Wem_h = cvt(em_hid_W, HH * HH);
    bf16 *Wem_o = cvt(em_out_W, PP * HH), *Wem_p = cvt(em_pj_W, HH * PP);
    bf16 *Wrh_i = cvt(rh_in_W, HH * PP), *Wrh_h = cvt(rh_hid_W, HH * HH);
    bf16 *Wrh_o = cvt(rh_out_W, CC * HH), *Wrh_p = cvt(rh_pj_W, HH * PP);
    bf16 *Wll[3], *Wlr[3];
    for (int i = 0; i < 3; ++i) { Wll[i] = cvt(cllW[i], CC * CC); Wlr[i] = cvt(clrW[i], CC * CC); }
    bf16 *W1i = cvt(h1_in_W, HH * 192), *W1h = cvt(h1_hid_W, HH * HH);
    bf16 *W1o = cvt(h1_out_W, HH * HH), *W1p = cvt(h1_pj_W, HH * 192);
    bf16 *W2i = cvt(h2_in_W, HH * HH), *W2h = cvt(h2_hid_W, HH * HH);

    // output slices (return order: node_emb, head(xy), head(yx), xy)
    float* out_ne  = (float*)d_out;
    float* out_hxy = out_ne + (size_t)NN * CC;
    float* out_hyx = out_hxy + QQ;
    float* out_xy  = out_hyx + QQ;

    // 1) embedder + point-sum
    k_embed<<<(NN * KK / 16) / EMB_WAVES, EMB_WAVES * 32, 0, stream>>>(
        x, Wem_i, Wem_h, Wem_o, Wem_p, em_in_b, em_hid_b, em_out_b, em_pj_b, hsumb);

    // 2) rho -> z0
    k_rho<<<(NN / 16 + RHO_WAVES - 1) / RHO_WAVES, RHO_WAVES * 32, 0, stream>>>(
        hsumb, Wrh_i, Wrh_h, Wrh_o, Wrh_p, rh_in_b, rh_hid_b, rh_out_b, rh_pj_b, z0);

    // 3) 3x SAGE residual convs (ping-pong z0/z1)
    float* zb[2] = {z0, z1};
    int cur = 0;
    for (int i = 0; i < 3; ++i) {
        k_fill<<<((long)NN * CC + 255) / 256, 256, 0, stream>>>(agg, (long)NN * CC, 0.0f);
        k_fill<<<(NN + 255) / 256, 256, 0, stream>>>(cnt, NN, 0.0f);
        k_edges<<<(unsigned)((EE * CC + 255) / 256), 256, 0, stream>>>(ei, zb[cur], agg, cnt);
        k_sage<<<(NN / 16 + SG_WAVES - 1) / SG_WAVES, SG_WAVES * 32, 0, stream>>>(
            zb[cur], agg, cnt, Wll[i], Wlr[i], cllB[i], zb[1 - cur]);
        cur = 1 - cur;
    }
    float* zf = zb[cur];

    // node embeddings out
    k_copy<<<((long)NN * CC + 255) / 256, 256, 0, stream>>>(zf, out_ne, (long)NN * CC);

    // 4) graph max pool
    k_fill<<<(GG * CC + 255) / 256, 256, 0, stream>>>(ge, GG * CC, -__builtin_huge_valf());
    k_gmax<<<((long)NN * CC + 255) / 256, 256, 0, stream>>>(zf, batch, ge);

    // 5) xy output
    k_xy<<<(unsigned)((QQ * 192 + 255) / 256), 256, 0, stream>>>(zf, ge, batch, srci, dsti, out_xy);

    // 6) heads (xy then yx, gather fused)
    const unsigned hblocks = (unsigned)((QQ / 16 + HD_WAVES - 1) / HD_WAVES);
    k_head<<<hblocks, HD_WAVES * 32, 0, stream>>>(
        zf, ge, batch, srci, dsti, 0, W1i, W1h, W1o, W1p, W2i, W2h,
        h1_in_b, h1_hid_b, h1_out_b, h1_pj_b, h2_in_b, h2_hid_b, h2_out_W, h2_out_b, out_hxy);
    k_head<<<hblocks, HD_WAVES * 32, 0, stream>>>(
        zf, ge, batch, srci, dsti, 1, W1i, W1h, W1o, W1p, W2i, W2h,
        h1_in_b, h1_hid_b, h1_out_b, h1_pj_b, h2_in_b, h2_hid_b, h2_out_W, h2_out_b, out_hyx);

    (void)in_sizes; (void)n_in; (void)out_size; (void)ws_size;
}
